// Embedding_27650999452107
// MI455X (gfx1250) — matrix operand order repaired and checked by reading, not compiled, not run on hardware
//
#include <hip/hip_runtime.h>

typedef _Float16 v16h __attribute__((ext_vector_type(16)));
typedef float    v8f  __attribute__((ext_vector_type(8)));

#define K_CODES 256
#define D_DIM   256
#define B_N     32
#define H_N     64
#define W_N     64
#define NPIX    (B_N * H_N * W_N)   // 131072 pixel vectors
#define NBLK    (B_N * H_N)         // 2048 blocks, 64 pixels each

// ---------------------------------------------------------------------------
// Prep: pack codebook into WMMA B-operand lane layout (f16) + squared norms.
// B fragment (kt: 16-code tile, dt: 32-d tile): lane L holds column
// n = L%16 (code kt*16+n), and 16 consecutive halves = d = dt*32 + (L<16?0:16)+j.
// Stored so a lane's 32B fragment slice is contiguous in memory.
// ---------------------------------------------------------------------------
__global__ void vq_prep(const float* __restrict__ weight,
                        _Float16* __restrict__ wsB,
                        float* __restrict__ wn) {
    const int k  = threadIdx.x;          // 0..255 (code index)
    const int kt = k >> 4;
    const int kl = k & 15;
    const float* wrow = weight + (size_t)k * D_DIM;
    float s = 0.0f;
    for (int d = 0; d < D_DIM; ++d) {
        const float v = wrow[d];
        s += v * v;
        const int dt   = d >> 5;
        const int r    = d & 31;
        const int lane = kl + ((r & 16) ? 16 : 0);
        const int j    = r & 15;
        wsB[((((size_t)kt * 8 + dt) * 32 + lane) << 4) + j] = (_Float16)v;
    }
    wn[k] = s;
}

// ---------------------------------------------------------------------------
// Main: one block per (b, h): 64 pixels (w=0..63) x D=256.
// 128 threads = 4 waves; wave wv owns pixels [wv*16, wv*16+16).
// ---------------------------------------------------------------------------
__global__ __launch_bounds__(128)
void vq_main(const float* __restrict__ z,
             const float* __restrict__ weight,
             const _Float16* __restrict__ wsB,
             const float* __restrict__ wn,
             float* __restrict__ partials,
             float* __restrict__ out_q,
             float* __restrict__ out_enc,
             float* __restrict__ out_idx) {
    extern __shared__ char smem_raw[];
    float* x_lds   = (float*)smem_raw;                   // [256][64] fp32, 64KB
    int*   idx_lds = (int*)(smem_raw + 65536);           // [64]
    float* red     = (float*)(smem_raw + 65536 + 256);   // [128]

    const int t    = threadIdx.x;
    const int lane = t & 31;
    const int wv   = t >> 5;
    const int blk  = blockIdx.x;
    const int b    = blk >> 6;
    const int h    = blk & 63;

    // ---- stage x slab into LDS as [d][w] (transpose happens via LDS) ----
    {
        const float* zbase = z + ((size_t)(b * D_DIM) * H_N + h) * W_N;
        #pragma unroll
        for (int it = 0; it < 2; ++it) {
            const int d = it * 128 + t;
            const float4* s4 = (const float4*)(zbase + (size_t)d * (H_N * W_N));
            float4* d4 = (float4*)(x_lds + d * 64);
            #pragma unroll
            for (int q = 0; q < 16; ++q) d4[q] = s4[q];
        }
    }
    __syncthreads();

    // ---- build A fragments (16 pixels x 32 d per tile), f32 -> f16 ----
    // A layout used here (contiguous k, as in wsB): lane L: row m = L%16; halves 0..7 hold
    // K = (L<16?0:16)+0..7, halves 8..15 hold K = (L<16?8:24)+0..7.
    const int mcol = (wv << 4) + (lane & 15);   // pixel (w) this lane covers
    const int hsel = lane >> 4;
    v16h afrag[8];
    #pragma unroll
    for (int dt = 0; dt < 8; ++dt) {
        const int dlo = dt * 32 + (hsel ? 16 : 0);
        const int dhi = dt * 32 + (hsel ? 24 : 8);
        v16h a;
        #pragma unroll
        for (int j = 0; j < 8; ++j) {
            a[j]     = (_Float16)x_lds[(dlo + j) * 64 + mcol];
            a[j + 8] = (_Float16)x_lds[(dhi + j) * 64 + mcol];
        }
        afrag[dt] = a;
    }

    // ---- 16 code tiles x 8 d tiles of WMMA, fused running argmin ----
    float bestv[8];
    int   besti[8];
    #pragma unroll
    for (int j = 0; j < 8; ++j) { bestv[j] = 3.0e38f; besti[j] = 0; }

    for (int kt = 0; kt < 16; ++kt) {
        v8f acc = {};
        const _Float16* bbase = wsB + (((size_t)kt * 8) * 32 + lane) * 16;
        #pragma unroll
        for (int dt = 0; dt < 8; ++dt) {
            const v16h bf = *(const v16h*)(bbase + (size_t)dt * 32 * 16);
            acc = __builtin_amdgcn_wmma_f32_16x16x32_f16(
                false, afrag[dt], false, bf, (short)0, acc, false, false);
        }
        // C/D layout: lane holds column n = lane%16; VGPR j = row j + hsel*8
        const int   kcol = (kt << 4) + (lane & 15);
        const float wnk  = wn[kcol];
        #pragma unroll
        for (int j = 0; j < 8; ++j) {
            const float sc = wnk - 2.0f * acc[j];
            if (sc < bestv[j] || (sc == bestv[j] && kcol < besti[j])) {
                bestv[j] = sc; besti[j] = kcol;
            }
        }
    }

    // ---- argmin reduction across the 16 lanes holding each row ----
    #pragma unroll
    for (int m = 1; m <= 8; m <<= 1) {
        #pragma unroll
        for (int j = 0; j < 8; ++j) {
            const float ov = __shfl_xor(bestv[j], m, 32);
            const int   oi = __shfl_xor(besti[j], m, 32);
            if (ov < bestv[j] || (ov == bestv[j] && oi < besti[j])) {
                bestv[j] = ov; besti[j] = oi;
            }
        }
    }
    if ((lane & 15) == 0) {
        #pragma unroll
        for (int j = 0; j < 8; ++j)
            idx_lds[(wv << 4) + (hsel << 3) + j] = besti[j];
    }
    __syncthreads();

    // ---- gather fp32 codebook rows; fused loss; overwrite x_lds in place ----
    float lsum = 0.0f;
    {
        const int p    = t & 63;      // pixel
        const int dh   = t >> 6;      // which 128-d half
        const int code = idx_lds[p];
        const float* wrow = weight + (size_t)code * D_DIM + dh * 128;
        for (int d2 = 0; d2 < 128; ++d2) {
            const int d  = dh * 128 + d2;
            const float qv = wrow[d2];
            const float xv = x_lds[d * 64 + p];
            const float df = qv - xv;
            lsum += df * df;
            x_lds[d * 64 + p] = qv;   // now holds quantized value
        }
    }
    __syncthreads();

    // ---- write quantized [B,D,H,W] (coalesced along w) ----
    {
        float* qbase = out_q + ((size_t)(b * D_DIM) * H_N + h) * W_N;
        for (int i = t; i < D_DIM * W_N; i += 128) {
            const int d = i >> 6;
            const int w = i & 63;
            qbase[(size_t)d * (H_N * W_N) + w] = x_lds[i];
        }
    }

    // ---- one-hot encodings + indices ----
    {
        const size_t nBase = (size_t)blk * 64;
        float* ebase = out_enc + nBase * K_CODES;
        for (int i = t; i < 64 * K_CODES; i += 128) {
            const int p = i >> 8;
            const int k = i & 255;
            ebase[i] = (k == idx_lds[p]) ? 1.0f : 0.0f;
        }
        if (t < 64) out_idx[nBase + t] = (float)idx_lds[t];
    }

    // ---- per-block loss partial (deterministic; no fp atomics) ----
    red[t] = lsum;
    __syncthreads();
    for (int s = 64; s > 0; s >>= 1) {
        if (t < s) red[t] += red[t + s];
        __syncthreads();
    }
    if (t == 0) partials[blk] = red[0];
}

// ---------------------------------------------------------------------------
// Finish: deterministic sum of 2048 partials; loss = vq + commit = 2 * mean.
// ---------------------------------------------------------------------------
__global__ void vq_finish(const float* __restrict__ partials,
                          float* __restrict__ out_loss) {
    __shared__ float red[256];
    float s = 0.0f;
    for (int i = threadIdx.x; i < NBLK; i += 256) s += partials[i];
    red[threadIdx.x] = s;
    __syncthreads();
    for (int st = 128; st > 0; st >>= 1) {
        if (threadIdx.x < st) red[threadIdx.x] += red[threadIdx.x + st];
        __syncthreads();
    }
    if (threadIdx.x == 0)
        out_loss[0] = 2.0f * red[0] / 33554432.0f;   // mean over B*H*W*D
}

extern "C" void kernel_launch(void* const* d_in, const int* in_sizes, int n_in,
                              void* d_out, int out_size, void* d_ws, size_t ws_size,
                              hipStream_t stream) {
    const float* z      = (const float*)d_in[0];   // [B, D, H, W]
    const float* weight = (const float*)d_in[1];   // [K, D]

    float* out      = (float*)d_out;
    float* out_loss = out;                                   // 1
    float* out_q    = out + 1;                               // B*D*H*W
    float* out_enc  = out_q + (size_t)B_N * D_DIM * H_N * W_N; // N*K
    float* out_idx  = out_enc + (size_t)NPIX * K_CODES;      // N

    _Float16* wsB    = (_Float16*)d_ws;                      // 131072 B
    float* wn        = (float*)((char*)d_ws + 131072);       // 1 KB
    float* partials  = (float*)((char*)d_ws + 132096);       // 8 KB

    vq_prep<<<1, 256, 0, stream>>>(weight, wsB, wn);

    const size_t smem = 65536 + 256 + 512;                   // x + idx + red
    vq_main<<<NBLK, 128, smem, stream>>>(z, weight, wsB, wn, partials,
                                         out_q, out_enc, out_idx);

    vq_finish<<<1, 256, 0, stream>>>(partials, out_loss);
}
